// OFTLinear_47132971106959
// MI455X (gfx1250) — matrix operand order repaired and checked by reading, not compiled, not run on hardware
//
#include <hip/hip_runtime.h>

// ---------------------------------------------------------------------------
// OFT linear for MI455X (gfx1250, wave32, WMMA).
// out = x @ G + b_eff, where G folds perm_in, block-diag(R_right), W^T,
// block-diag(R_left), perm_out into a single 2048x2048 matrix built once
// per call (~0.5 GFLOP), leaving one WMMA GEMM as the hot loop.
// Precision: split-bf16 (hi+lo) with f32 accumulation -> ~1e-5 rel error.
// ---------------------------------------------------------------------------

#define IN_F   2048
#define OUT_F  2048
#define BSZ    32
#define RBLK   64          // 2048 / 32 blocks per side
#define NELEM  496         // 32*31/2
#define MTOT   32768       // 4 * 8192 rows

typedef __attribute__((ext_vector_type(16))) __bf16 v16bf;
typedef __attribute__((ext_vector_type(8)))  float  v8f;

union FragU { uint4 u[2]; v16bf v; };

__device__ __forceinline__ unsigned short bf16_rn(float f) {
  unsigned u = __float_as_uint(f);
  u += 0x7FFFu + ((u >> 16) & 1u);           // round-to-nearest-even
  return (unsigned short)(u >> 16);
}
__device__ __forceinline__ void bf16_split(float f, unsigned short& h, unsigned short& l) {
  h = bf16_rn(f);
  l = bf16_rn(f - __uint_as_float(((unsigned)h) << 16));
}

// ---------------------------------------------------------------------------
// Kernel A: build R_all[128][32][32] = I + 2Q + 2Q^2 + 2Q^3 + 2Q^4
//   blocks 0..63   from oft_L  (R_left)
//   blocks 64..127 from oft_R  (R_right)
// ---------------------------------------------------------------------------
__global__ __launch_bounds__(256) void build_R_kernel(
    const float* __restrict__ oft_L, const float* __restrict__ oft_R,
    const int* __restrict__ rows, const int* __restrict__ cols,
    float* __restrict__ R_all) {
  __shared__ float Q[32][33];
  __shared__ float P[32][33];
  const int blk = blockIdx.x;
  const int tid = threadIdx.x;

  #pragma unroll
  for (int s = 0; s < 4; ++s) {
    int e = tid + 256 * s;
    Q[e >> 5][e & 31] = 0.0f;
  }
  __syncthreads();
  #pragma unroll
  for (int s = 0; s < 2; ++s) {
    int e = tid + 256 * s;
    if (e < NELEM) {
      float v = (blk < RBLK) ? oft_L[blk * NELEM + e]
                             : oft_R[(blk - RBLK) * NELEM + e];
      int r = rows[e], c = cols[e];
      Q[r][c] = v;
      Q[c][r] = -v;
    }
  }
  __syncthreads();

  float racc[4];
  int ei[4], ej[4];
  #pragma unroll
  for (int s = 0; s < 4; ++s) {
    int e = tid + 256 * s;
    ei[s] = e >> 5; ej[s] = e & 31;
    float q = Q[ei[s]][ej[s]];
    racc[s] = (ei[s] == ej[s] ? 1.0f : 0.0f) + 2.0f * q;
    P[ei[s]][ej[s]] = q;
  }
  __syncthreads();

  for (int p = 2; p <= 4; ++p) {        // powers Q^2, Q^3, Q^4
    float tmp[4];
    #pragma unroll
    for (int s = 0; s < 4; ++s) {
      float acc = 0.0f;
      int i = ei[s], j = ej[s];
      #pragma unroll
      for (int k = 0; k < 32; ++k) acc += P[i][k] * Q[k][j];
      tmp[s] = acc;
    }
    __syncthreads();
    #pragma unroll
    for (int s = 0; s < 4; ++s) {
      P[ei[s]][ej[s]] = tmp[s];
      racc[s] += 2.0f * tmp[s];
    }
    __syncthreads();
  }
  #pragma unroll
  for (int s = 0; s < 4; ++s)
    R_all[blk * 1024 + ei[s] * 32 + ej[s]] = racc[s];
}

// ---------------------------------------------------------------------------
// Kernel B: per (r,l) block pair, H = R_right[r] * W_blk(l,r)^T * R_left[l];
// scatter through perm_in (rows=GEMM-K) / perm_out (cols=GEMM-N) into
// transposed Gt[n][k] stored as bf16 hi/lo. Also builds b_eff.
// ---------------------------------------------------------------------------
__global__ __launch_bounds__(256) void build_G_kernel(
    const float* __restrict__ W, const float* __restrict__ b,
    const int* __restrict__ perm_in, const int* __restrict__ perm_out,
    const float* __restrict__ R_all,
    unsigned short* __restrict__ Gt_hi, unsigned short* __restrict__ Gt_lo,
    float* __restrict__ b_eff) {
  __shared__ float Wb[32][33], Rl[32][33], Rr[32][33], U[32][33];
  const int r = blockIdx.x;   // input 32-block
  const int l = blockIdx.y;   // output 32-block
  const int tid = threadIdx.x;

  #pragma unroll
  for (int s = 0; s < 4; ++s) {
    int e = tid + 256 * s;
    int i = e >> 5, j = e & 31;
    Wb[i][j] = W[(size_t)(l * 32 + i) * IN_F + r * 32 + j];
    Rl[i][j] = R_all[l * 1024 + i * 32 + j];
    Rr[i][j] = R_all[(RBLK + r) * 1024 + i * 32 + j];
  }
  __syncthreads();
  // U[jj][cc] = sum_kk Wb[kk][jj] * Rl[kk][cc]   (= W_blk^T @ R_left)
  #pragma unroll
  for (int s = 0; s < 4; ++s) {
    int e = tid + 256 * s;
    int jj = e >> 5, cc = e & 31;
    float acc = 0.0f;
    #pragma unroll
    for (int kk = 0; kk < 32; ++kk) acc += Wb[kk][jj] * Rl[kk][cc];
    U[jj][cc] = acc;
  }
  __syncthreads();
  // H[km][cc] = sum_jj Rr[km][jj] * U[jj][cc], scatter into Gt (transposed)
  #pragma unroll
  for (int s = 0; s < 4; ++s) {
    int e = tid + 256 * s;
    int km = e >> 5, cc = e & 31;
    float acc = 0.0f;
    #pragma unroll
    for (int jj = 0; jj < 32; ++jj) acc += Rr[km][jj] * U[jj][cc];
    int ig = perm_in[r * 32 + km];      // GEMM K index
    int cg = perm_out[l * 32 + cc];     // GEMM N index
    unsigned short h, lo;
    bf16_split(acc, h, lo);
    Gt_hi[(size_t)cg * IN_F + ig] = h;
    Gt_lo[(size_t)cg * IN_F + ig] = lo;
  }
  if (r == 0 && tid < 32) {
    float acc = 0.0f;
    #pragma unroll
    for (int kk = 0; kk < 32; ++kk) acc += Rl[kk][tid] * b[l * 32 + kk];
    b_eff[perm_out[l * 32 + tid]] = acc;
  }
}

// ---------------------------------------------------------------------------
// Kernel C: out[32768,2048] = x @ G + b_eff via v_wmma_f32_16x16x32_bf16.
// 128x128 tile / 256 threads (8 wave32s, 2x4 wave grid, 4x2 16x16 tiles
// per wave). Split-bf16: hi*hi + hi*lo + lo*hi (lo*lo dropped).
// ---------------------------------------------------------------------------
__global__ __launch_bounds__(256) void gemm_kernel(
    const float* __restrict__ x,
    const unsigned short* __restrict__ Gt_hi,
    const unsigned short* __restrict__ Gt_lo,
    const float* __restrict__ b_eff,
    float* __restrict__ out) {
  __shared__ unsigned short xs_hi[128][32];
  __shared__ unsigned short xs_lo[128][32];
  __shared__ unsigned short gs_hi[128][32];   // [n][k] (pre-transposed in ws)
  __shared__ unsigned short gs_lo[128][32];

  const int tid  = threadIdx.x;
  const int lane = tid & 31;
  const int wave = tid >> 5;
  const int wm   = wave >> 2;        // 0..1 : 64-row half
  const int wn   = wave & 3;         // 0..3 : 32-col quarter
  const int ln15 = lane & 15;
  const int lhi  = lane >> 4;        // lane half (ISA fragment layouts)

  const int m0 = blockIdx.y * 128;
  const int n0 = blockIdx.x * 128;

  v8f acc[4][2] = {};

  for (int k0 = 0; k0 < IN_F; k0 += 32) {
    __syncthreads();
    // ---- stage x tile (128x32 f32) -> bf16 hi/lo in LDS ----
    #pragma unroll
    for (int it = 0; it < 4; ++it) {
      int idx = tid + 256 * it;
      int row = idx >> 3, c4 = (idx & 7) * 4;
      const float4 v = *(const float4*)(x + (size_t)(m0 + row) * IN_F + k0 + c4);
      unsigned short h, l;
      bf16_split(v.x, h, l); xs_hi[row][c4 + 0] = h; xs_lo[row][c4 + 0] = l;
      bf16_split(v.y, h, l); xs_hi[row][c4 + 1] = h; xs_lo[row][c4 + 1] = l;
      bf16_split(v.z, h, l); xs_hi[row][c4 + 2] = h; xs_lo[row][c4 + 2] = l;
      bf16_split(v.w, h, l); xs_hi[row][c4 + 3] = h; xs_lo[row][c4 + 3] = l;
    }
    // ---- stage G tiles (already bf16, [n][k]) ----
    #pragma unroll
    for (int it = 0; it < 2; ++it) {
      int idx = tid + 256 * it;
      int row = idx >> 2, c8 = (idx & 3) * 8;
      *(uint4*)&gs_hi[row][c8] = *(const uint4*)(Gt_hi + (size_t)(n0 + row) * IN_F + k0 + c8);
      *(uint4*)&gs_lo[row][c8] = *(const uint4*)(Gt_lo + (size_t)(n0 + row) * IN_F + k0 + c8);
    }
    if (k0 + 32 < IN_F)   // hint next x panel (global_prefetch_b8)
      __builtin_prefetch(x + (size_t)(m0 + (tid >> 3)) * IN_F + k0 + 32, 0, 1);
    __syncthreads();

    // ---- load fragments per ISA 16-bit layouts ----
    FragU a_hi[4], a_lo[4], b_hi[2], b_lo[2];
    const int kbA = lhi * 8;           // A: lanes0-15 K0-7/16-23; lanes16-31 K8-15/24-31
    #pragma unroll
    for (int mt = 0; mt < 4; ++mt) {
      int m = wm * 64 + mt * 16 + ln15;
      a_hi[mt].u[0] = *(const uint4*)&xs_hi[m][kbA];
      a_hi[mt].u[1] = *(const uint4*)&xs_hi[m][kbA + 16];
      a_lo[mt].u[0] = *(const uint4*)&xs_lo[m][kbA];
      a_lo[mt].u[1] = *(const uint4*)&xs_lo[m][kbA + 16];
    }
    const int kbB = lhi * 8;           // B: lanes0-15 K0-7/16-23; lanes16-31 K8-15/24-31
    #pragma unroll
    for (int nt = 0; nt < 2; ++nt) {
      int n = wn * 32 + nt * 16 + ln15;
      b_hi[nt].u[0] = *(const uint4*)&gs_hi[n][kbB];
      b_hi[nt].u[1] = *(const uint4*)&gs_hi[n][kbB + 16];
      b_lo[nt].u[0] = *(const uint4*)&gs_lo[n][kbB];
      b_lo[nt].u[1] = *(const uint4*)&gs_lo[n][kbB + 16];
    }

    // ---- 24 WMMA per wave per K-step ----
    #pragma unroll
    for (int mt = 0; mt < 4; ++mt) {
      #pragma unroll
      for (int nt = 0; nt < 2; ++nt) {
        acc[mt][nt] = __builtin_amdgcn_wmma_f32_16x16x32_bf16(
            false, a_hi[mt].v, false, b_hi[nt].v, (short)0, acc[mt][nt], false, false);
        acc[mt][nt] = __builtin_amdgcn_wmma_f32_16x16x32_bf16(
            false, a_hi[mt].v, false, b_lo[nt].v, (short)0, acc[mt][nt], false, false);
        acc[mt][nt] = __builtin_amdgcn_wmma_f32_16x16x32_bf16(
            false, a_lo[mt].v, false, b_hi[nt].v, (short)0, acc[mt][nt], false, false);
      }
    }
  }

  // ---- epilogue: C layout lane(0-15)=N, vgpr v -> M = lhi*8 + v ----
  #pragma unroll
  for (int nt = 0; nt < 2; ++nt) {
    int col = n0 + wn * 32 + nt * 16 + ln15;
    float bias = b_eff[col];
    #pragma unroll
    for (int mt = 0; mt < 4; ++mt) {
      int rbase = m0 + wm * 64 + mt * 16 + lhi * 8;
      #pragma unroll
      for (int v = 0; v < 8; ++v)
        out[(size_t)(rbase + v) * OUT_F + col] = acc[mt][nt][v] + bias;
    }
  }
}

// ---------------------------------------------------------------------------
extern "C" void kernel_launch(void* const* d_in, const int* in_sizes, int n_in,
                              void* d_out, int out_size, void* d_ws, size_t ws_size,
                              hipStream_t stream) {
  const float* x       = (const float*)d_in[0];
  const float* oft_L   = (const float*)d_in[1];
  const float* oft_R   = (const float*)d_in[2];
  const float* W       = (const float*)d_in[3];
  const float* b       = (const float*)d_in[4];
  const int*   rows    = (const int*)d_in[5];
  const int*   cols    = (const int*)d_in[6];
  const int*   perm_in = (const int*)d_in[7];
  const int*   perm_out= (const int*)d_in[9];
  float* out = (float*)d_out;

  // workspace layout: R_all (512KB) | Gt_hi (8MB) | Gt_lo (8MB) | b_eff (8KB)
  char* ws = (char*)d_ws;
  float*          R_all = (float*)ws;
  unsigned short* Gt_hi = (unsigned short*)(ws + (512 << 10));
  unsigned short* Gt_lo = (unsigned short*)(ws + (512 << 10) + (8 << 20));
  float*          b_eff = (float*)(ws + (512 << 10) + (16 << 20));

  hipLaunchKernelGGL(build_R_kernel, dim3(128), dim3(256), 0, stream,
                     oft_L, oft_R, rows, cols, R_all);
  hipLaunchKernelGGL(build_G_kernel, dim3(RBLK, RBLK), dim3(256), 0, stream,
                     W, b, perm_in, perm_out, R_all, Gt_hi, Gt_lo, b_eff);
  hipLaunchKernelGGL(gemm_kernel, dim3(OUT_F / 128, MTOT / 128), dim3(256), 0, stream,
                     x, Gt_hi, Gt_lo, b_eff, out);

  (void)in_sizes; (void)n_in; (void)out_size; (void)ws_size;
}
